// Swin3DBlock_76785425318613
// MI455X (gfx1250) — compile-verified, matrix operand order checked by reading, not run on hardware
//
#include <hip/hip_runtime.h>
#include <hip/hip_bf16.h>

// ---------------------------------------------------------------------------
// Swin3D block pair for MI455X (gfx1250, wave32, WMMA f32_16x16x32_f16).
// One workgroup (256 threads = 8 waves) per 4x4x4 window (T=64 tokens, D=384).
// All WMMA B operands are K-contiguous: weights are transposed to [N][K] f16
// in d_ws at conversion time, V is stored transposed per head, so every
// fragment load is two contiguous 16B loads (ds/global _b128).
// Per-head rel-pos bias (8KB f16) is DMA'd into LDS by the Tensor Data Mover
// (tensor_load_to_lds + s_wait_tensorcnt) and consumed as the WMMA C operand.
// Both rolls of the shifted layer are folded into load/store index math.
// ---------------------------------------------------------------------------

typedef _Float16 v16h __attribute__((ext_vector_type(16)));
typedef float    v8f  __attribute__((ext_vector_type(8)));
typedef unsigned int v4u __attribute__((ext_vector_type(4)));
typedef int      v8i  __attribute__((ext_vector_type(8)));
typedef int      v4i  __attribute__((ext_vector_type(4)));

#ifndef __has_builtin
#define __has_builtin(x) 0
#endif
#if __has_builtin(__builtin_amdgcn_tensor_load_to_lds) && \
    __has_builtin(__builtin_amdgcn_s_wait_tensorcnt)
#define USE_TDM 1
#else
#define USE_TDM 0
#endif

#define DD   384
#define TT   64
#define HH   12
#define DHD  32
#define SH   392   // row stride (halves) for 64x384 f16 LDS buffers
#define SQK  40    // row stride for q/k (64x32)
#define SVT  72    // row stride for v^T (32x64)
#define SS   72    // row stride for scores / attn (64x64)
#define SG   136   // row stride for gelu chunk (64x128)
#define FCH  128   // MLP hidden chunk

static __device__ inline v8f wmma32(v16h a, v16h b, v8f c) {
  // D = A(16x32 f16) * B(32x16 f16) + C(16x16 f32)
  return __builtin_amdgcn_wmma_f32_16x16x32_f16(false, a, false, b, (short)0, c,
                                                false, false);
}

// A fragment (16x32) from row-major [M][K] f16 buffer. lane%16 = M row;
// lane<16 holds K 0-7 / 16-23, lane>=16 holds K 8-15 / 24-31.
static __device__ inline v16h load_a_rm(const _Float16* src, int row0, int col0,
                                        int stride, int lane) {
  v16h a;
  const int m  = lane & 15;
  const int kb = (lane >> 4) << 3;
  const _Float16* p = src + (size_t)(row0 + m) * stride + col0 + kb;
#pragma unroll
  for (int j = 0; j < 8; ++j) { a[j] = p[j]; a[8 + j] = p[16 + j]; }
  return a;
}

// B fragment (32x16) from K-contiguous [N][K] f16 buffer (LDS or global):
// lane%16 = N column, lane<16 holds K 0-7 / 16-23, lane>=16 K 8-15 / 24-31 (two 16B loads).
static __device__ inline v16h load_b_nk(const _Float16* src, int k0, int n0,
                                        int stride, int lane) {
  v16h b;
  const int n  = lane & 15;
  const int kb = (lane >> 4) << 3;
  const _Float16* p = src + (size_t)(n0 + n) * stride + k0 + kb;
#pragma unroll
  for (int j = 0; j < 8; ++j) { b[j] = p[j]; b[8 + j] = p[16 + j]; }
  return b;
}

// C/D fragment helpers: vgpr r -> M = (lane/16)*8 + r, N = lane%16.
static __device__ inline v8f load_c_f16(const _Float16* src, int row0, int col0,
                                        int stride, int lane) {
  v8f c;
  const int n  = lane & 15;
  const int mb = (lane >> 4) << 3;
#pragma unroll
  for (int r = 0; r < 8; ++r)
    c[r] = (float)src[(size_t)(row0 + mb + r) * stride + col0 + n];
  return c;
}

static __device__ inline void store_d_f16(_Float16* dst, v8f c, int row0,
                                          int col0, int stride, int lane,
                                          float scale) {
  const int n  = lane & 15;
  const int mb = (lane >> 4) << 3;
#pragma unroll
  for (int r = 0; r < 8; ++r)
    dst[(size_t)(row0 + mb + r) * stride + col0 + n] = (_Float16)(c[r] * scale);
}

// Transposed D store: element (m,n) -> dst[n][m] (used to build v^T).
static __device__ inline void store_d_f16_tr(_Float16* dst, v8f c, int row0,
                                             int col0, int stride, int lane) {
  const int n  = lane & 15;
  const int mb = (lane >> 4) << 3;
#pragma unroll
  for (int r = 0; r < 8; ++r)
    dst[(size_t)(col0 + n) * stride + row0 + mb + r] = (_Float16)c[r];
}

static __device__ inline void store_d_f32(float* dst, v8f c, int row0, int col0,
                                          int stride, int lane) {
  const int n  = lane & 15;
  const int mb = (lane >> 4) << 3;
#pragma unroll
  for (int r = 0; r < 8; ++r)
    dst[(size_t)(row0 + mb + r) * stride + col0 + n] = c[r];
}

// LayerNorm over D=384 per token; src/dst are SH-strided f16 LDS buffers
// (may alias). 4 threads cooperate per token.
static __device__ inline void layernorm_lds(_Float16* dst, const _Float16* src,
                                            const float* g, const float* b,
                                            float* s_red, float* s_m,
                                            float* s_r, int tid) {
  const int t = tid >> 2, p = tid & 3;
  float s = 0.f, s2 = 0.f;
  const _Float16* row = src + (size_t)t * SH + p * 96;
#pragma unroll 4
  for (int j = 0; j < 96; ++j) { float v = (float)row[j]; s += v; s2 += v * v; }
  s_red[tid] = s;
  s_red[256 + tid] = s2;
  __syncthreads();
  if (p == 0) {
    float ss = 0.f, qq = 0.f;
#pragma unroll
    for (int j = 0; j < 4; ++j) { ss += s_red[t * 4 + j]; qq += s_red[256 + t * 4 + j]; }
    float m   = ss * (1.0f / 384.0f);
    float var = qq * (1.0f / 384.0f) - m * m;
    s_m[t] = m;
    s_r[t] = rsqrtf(var + 1e-5f);
  }
  __syncthreads();
  for (int idx = tid; idx < TT * DD; idx += 256) {
    int t2 = idx / DD, d = idx - t2 * DD;
    float v = ((float)src[(size_t)t2 * SH + d] - s_m[t2]) * s_r[t2] * g[d] + b[d];
    dst[(size_t)t2 * SH + d] = (_Float16)v;
  }
  __syncthreads();
}

__global__ __launch_bounds__(256, 1)
void swin3d_block_kernel(const float* __restrict__ x_in,
                         float* __restrict__ x_out,
                         const float* __restrict__ ln1_g, const float* __restrict__ ln1_b,
                         const float* __restrict__ qkv_b, const float* __restrict__ proj_b,
                         const float* __restrict__ ln2_g, const float* __restrict__ ln2_b,
                         const float* __restrict__ fc1_b, const float* __restrict__ fc2_b,
                         const _Float16* __restrict__ wqkvT,  // [1152][384]
                         const _Float16* __restrict__ wprojT, // [384][384] (W^T)
                         const _Float16* __restrict__ w1T,    // [1536][384]
                         const _Float16* __restrict__ w2T,    // [384][1536]
                         const _Float16* __restrict__ bias16, // [12][64][64] f16
                         int shift) {
  __shared__ _Float16 s_h[TT * SH];                 // LN1/LN2 output
  __shared__ _Float16 s_o[TT * SH];                 // attn output / gelu chunk
  __shared__ __align__(16) unsigned char s_u[TT * SH * 2]; // attn scratch | xw
  __shared__ float s_red[512];
  __shared__ float s_m[64];
  __shared__ float s_r[64];

  _Float16* s_q  = (_Float16*)s_u;                         // 64 x SQK
  _Float16* s_k  = s_q + TT * SQK;                         // 64 x SQK
  _Float16* s_vt = s_k + TT * SQK;                         // 32 x SVT (v^T)
  float*    s_s  = (float*)(s_u + ((size_t)2 * TT * SQK + 32 * SVT) * sizeof(_Float16));
  _Float16* s_a  = (_Float16*)((unsigned char*)s_s + (size_t)TT * SS * sizeof(float));
  _Float16* s_bias = s_a;    // bias staged where attn probs go later (disjoint lifetimes)
  _Float16* s_xw = (_Float16*)s_u;                         // reuses attn scratch region

  const int tid  = threadIdx.x;
  const int lane = tid & 31;
  const int wv   = tid >> 5;
  const int blk  = blockIdx.x;
  const int wxi = blk & 15, wyi = (blk >> 4) & 15, wzi = blk >> 8;

  auto gaddr = [&](int t, int d) -> size_t {
    int lz = t >> 4, ly = (t >> 2) & 3, lx = t & 3;
    int sz = (wzi * 4 + lz - shift + 32) & 31;
    int sy = (wyi * 4 + ly - shift + 64) & 63;
    int sx = (wxi * 4 + lx - shift + 64) & 63;
    return ((size_t)((sz * 64 + sy) * 64 + sx)) * DD + d;
  };

  // ---- load window + LN1 -> s_h ----
  for (int idx = tid; idx < TT * DD; idx += 256) {
    int t = idx / DD, d = idx - t * DD;
    s_h[(size_t)t * SH + d] = (_Float16)x_in[gaddr(t, d)];
  }
  __syncthreads();
  layernorm_lds(s_h, s_h, ln1_g, ln1_b, s_red, s_m, s_r, tid);

  // ---- attention, per head ----
  const float scale = 0.17677669529663687f; // 32^-0.5
  for (int h = 0; h < HH; ++h) {
#if USE_TDM
    // TDM: DMA this head's 8KB f16 bias tile into LDS while QKV GEMMs run.
    if (wv == 0) {
      unsigned lds_off = (unsigned)(size_t)s_bias;  // low 32b of LDS aperture addr
      unsigned long long ga =
          (unsigned long long)(size_t)(bias16 + (size_t)h * (TT * TT));
      v4u g0;
      g0[0] = 1u;                                   // count=1 (valid, user mode)
      g0[1] = lds_off;                              // lds_addr
      g0[2] = (unsigned)(ga & 0xffffffffu);         // global_addr[31:0]
      g0[3] = (unsigned)((ga >> 32) & 0x01ffffffu)  // global_addr[56:32]
              | (2u << 30);                         // type = 2 ("image")
      v8i g1;
      g1[0] = 0x00010000;   // workgroup_mask=0, data_size=1 (2 bytes)
      g1[1] = 0x10000000;   // tensor_dim0 = 4096 (bits 79:48, low half here)
      g1[2] = 0x00010000;   // tensor_dim0 hi = 0, tensor_dim1 = 1
      g1[3] = 0x10000000;   // tensor_dim1 hi = 0, tile_dim0 = 4096
      g1[4] = 0;            // tile_dim1 = 0 (unused), tile_dim2 = 0
      g1[5] = 4096;         // tensor_dim0_stride (low 32)
      g1[6] = 0;
      g1[7] = 0;
      v4i gz4 = {0, 0, 0, 0};                       // D# groups 2/3 unused (<=2D)
      v8i gz8 = {0, 0, 0, 0, 0, 0, 0, 0};           // extra VADDR group unused
      __builtin_amdgcn_tensor_load_to_lds(g0, g1, gz4, gz4, gz8, 0);
    }
#endif
    // q/k/v = LN(x) @ Wqkv[:, head cols] + b   (q pre-scaled, v stored ^T)
    for (int tile = wv; tile < 24; tile += 8) {
      int mat = tile >> 3;          // 0=q, 1=k, 2=v
      int rr  = tile & 7;
      int mt = rr >> 1, nt = rr & 1;
      int col = mat * DD + h * DHD + nt * 16;
      float bc = qkv_b[col + (lane & 15)];
      v8f c;
#pragma unroll
      for (int i = 0; i < 8; ++i) c[i] = bc;
      for (int k = 0; k < 12; ++k) {
        __builtin_prefetch(wqkvT + (size_t)(col + (lane & 15)) * DD + k * 32 + 32, 0, 0);
        v16h a = load_a_rm(s_h, mt * 16, k * 32, SH, lane);
        v16h b = load_b_nk(wqkvT, k * 32, col, DD, lane);
        c = wmma32(a, b, c);
      }
      if (mat == 0)      store_d_f16(s_q, c, mt * 16, nt * 16, SQK, lane, scale);
      else if (mat == 1) store_d_f16(s_k, c, mt * 16, nt * 16, SQK, lane, 1.0f);
      else               store_d_f16_tr(s_vt, c, mt * 16, nt * 16, SVT, lane);
    }
#if USE_TDM
    if (wv == 0) __builtin_amdgcn_s_wait_tensorcnt(0);
#endif
    __syncthreads();

    // scores = q @ k^T + rel-pos bias (bias as C operand), K = dh = 32
#if USE_TDM
    const _Float16* bsrc = s_bias;
#else
    const _Float16* bsrc = bias16 + (size_t)h * (TT * TT);
#endif
    for (int tile = wv; tile < 16; tile += 8) {
      int mt = tile >> 2, nt = tile & 3;
      v8f c = load_c_f16(bsrc, mt * 16, nt * 16, 64, lane);
      v16h a = load_a_rm(s_q, mt * 16, 0, SQK, lane);
      v16h b = load_b_nk(s_k, 0, nt * 16, SQK, lane);  // k is [s][d] = B^T source
      c = wmma32(a, b, c);
      store_d_f32(s_s, c, mt * 16, nt * 16, SS, lane);
    }
    __syncthreads();

    // softmax over rows of s_s -> s_a (f16)
    {
      const int t = tid >> 2, p = tid & 3;
      float* row = s_s + (size_t)t * SS + p * 16;
      float mx = -1e30f;
#pragma unroll
      for (int j = 0; j < 16; ++j) mx = fmaxf(mx, row[j]);
      s_red[tid] = mx;
      __syncthreads();
      float rmax = fmaxf(fmaxf(s_red[t * 4], s_red[t * 4 + 1]),
                         fmaxf(s_red[t * 4 + 2], s_red[t * 4 + 3]));
      float sum = 0.f;
      _Float16* arow = s_a + (size_t)t * SS + p * 16;
#pragma unroll
      for (int j = 0; j < 16; ++j) {
        float e = __expf(row[j] - rmax);
        arow[j] = (_Float16)e;
        sum += e;
      }
      s_red[256 + tid] = sum;
      __syncthreads();
      float rinv = 1.0f / (s_red[256 + t * 4] + s_red[256 + t * 4 + 1] +
                           s_red[256 + t * 4 + 2] + s_red[256 + t * 4 + 3]);
#pragma unroll
      for (int j = 0; j < 16; ++j) arow[j] = (_Float16)((float)arow[j] * rinv);
    }
    __syncthreads();

    // o = attn @ v  (K = 64 -> 2 steps); one 16x16 tile per wave
    {
      int mt = wv >> 1, nt = wv & 1;
      v8f c = {};
#pragma unroll
      for (int k = 0; k < 2; ++k) {
        v16h a = load_a_rm(s_a, mt * 16, k * 32, SS, lane);
        v16h b = load_b_nk(s_vt, k * 32, nt * 16, SVT, lane);  // v^T is [d][s]
        c = wmma32(a, b, c);
      }
      store_d_f16(s_o, c, mt * 16, h * DHD + nt * 16, SH, lane, 1.0f);
    }
    __syncthreads();
  }

  // ---- proj + residual: xw = x + proj_b + o @ Wproj (accumulate in place) ----
  for (int idx = tid; idx < TT * DD; idx += 256) {
    int t = idx / DD, d = idx - t * DD;
    s_xw[(size_t)t * SH + d] = (_Float16)(x_in[gaddr(t, d)] + proj_b[d]);
  }
  __syncthreads();
  for (int i = 0; i < 12; ++i) {
    int tile = wv * 12 + i;
    int mt = tile / 24, nt = tile % 24;
    v8f c = load_c_f16(s_xw, mt * 16, nt * 16, SH, lane);
    for (int k = 0; k < 12; ++k) {
      v16h a = load_a_rm(s_o, mt * 16, k * 32, SH, lane);
      v16h b = load_b_nk(wprojT, k * 32, nt * 16, DD, lane);
      c = wmma32(a, b, c);
    }
    store_d_f16(s_xw, c, mt * 16, nt * 16, SH, lane, 1.0f);
  }
  __syncthreads();

  // ---- LN2 -> s_h; pre-add fc2 bias into the residual ----
  layernorm_lds(s_h, s_xw, ln2_g, ln2_b, s_red, s_m, s_r, tid);
  for (int idx = tid; idx < TT * DD; idx += 256) {
    int t = idx / DD, d = idx - t * DD;
    s_xw[(size_t)t * SH + d] =
        (_Float16)((float)s_xw[(size_t)t * SH + d] + fc2_b[d]);
  }
  __syncthreads();

  // ---- MLP, chunked over hidden dim (1536 = 12 x 128) ----
  _Float16* s_g = s_o; // reuse attn-output buffer for gelu chunk (64x136)
  for (int ch = 0; ch < 12; ++ch) {
    const int colbase = ch * FCH;
    // g = gelu(LN2 @ W1[:, chunk] + b1)
    for (int tile = wv; tile < 32; tile += 8) {
      int mt = tile >> 3, nt = tile & 7;
      int col = colbase + nt * 16;
      float bc = fc1_b[col + (lane & 15)];
      v8f c;
#pragma unroll
      for (int i = 0; i < 8; ++i) c[i] = bc;
      for (int k = 0; k < 12; ++k) {
        __builtin_prefetch(w1T + (size_t)(col + (lane & 15)) * DD + k * 32 + 32, 0, 0);
        v16h a = load_a_rm(s_h, mt * 16, k * 32, SH, lane);
        v16h b = load_b_nk(w1T, k * 32, col, DD, lane);
        c = wmma32(a, b, c);
      }
      const int n = lane & 15, mb = (lane >> 4) << 3;
#pragma unroll
      for (int r = 0; r < 8; ++r) {
        float xv = c[r];
        float gl = 0.5f * xv *
                   (1.0f + tanhf(0.7978845608f * (xv + 0.044715f * xv * xv * xv)));
        s_g[(size_t)(mt * 16 + mb + r) * SG + nt * 16 + n] = (_Float16)gl;
      }
    }
    __syncthreads();
    // xw += g @ W2[chunk, :]
    for (int i = 0; i < 12; ++i) {
      int tile = wv * 12 + i;
      int mt = tile / 24, nt = tile % 24;
      v8f c = load_c_f16(s_xw, mt * 16, nt * 16, SH, lane);
#pragma unroll
      for (int k = 0; k < 4; ++k) {
        v16h a = load_a_rm(s_g, mt * 16, k * 32, SG, lane);
        v16h b = load_b_nk(w2T, colbase + k * 32, nt * 16, 4 * DD, lane);
        c = wmma32(a, b, c);
      }
      store_d_f16(s_xw, c, mt * 16, nt * 16, SH, lane, 1.0f);
    }
    __syncthreads();
  }

  // ---- write back (fold un-roll for shifted layer) ----
  for (int idx = tid; idx < TT * DD; idx += 256) {
    int t = idx / DD, d = idx - t * DD;
    x_out[gaddr(t, d)] = (float)s_xw[(size_t)t * SH + d];
  }
}

// ---------------------------------------------------------------------------
// Convert f32 [K][N] weight -> f16 [N][K] (K-contiguous for WMMA B loads).
__global__ void transpose_cvt_kernel(const float* __restrict__ src,
                                     _Float16* __restrict__ dst, int K, int N) {
  int i = blockIdx.x * 256 + threadIdx.x;
  if (i < K * N) {
    int n = i / K, k = i - n * K;
    dst[i] = (_Float16)src[(size_t)k * N + n];
  }
}

__global__ void bias_expand_kernel(const float* __restrict__ rpb,
                                   const int* __restrict__ rel_idx,
                                   _Float16* __restrict__ out) {
  int i = blockIdx.x * 256 + threadIdx.x;
  if (i < HH * TT * TT) {
    int h = i / (TT * TT);
    int ts = i - h * (TT * TT);
    out[i] = (_Float16)rpb[rel_idx[ts] * HH + h];
  }
}

// ---------------------------------------------------------------------------
extern "C" void kernel_launch(void* const* d_in, const int* in_sizes, int n_in,
                              void* d_out, int out_size, void* d_ws,
                              size_t ws_size, hipStream_t stream) {
  (void)in_sizes; (void)n_in; (void)ws_size;

  // setup_inputs dict order; params dicts flattened as jax pytree leaves
  // (alphabetical key order): fc1_b, fc1_w, fc2_b, fc2_w, ln1_b, ln1_g,
  // ln2_b, ln2_g, proj_b, proj_w, qkv_b, qkv_w, rpb_table.
  const float* x_in = (const float*)d_in[0];

  const float* w_fc1_b  = (const float*)d_in[1];
  const float* w_fc1_w  = (const float*)d_in[2];
  const float* w_fc2_b  = (const float*)d_in[3];
  const float* w_fc2_w  = (const float*)d_in[4];
  const float* w_ln1_b  = (const float*)d_in[5];
  const float* w_ln1_g  = (const float*)d_in[6];
  const float* w_ln2_b  = (const float*)d_in[7];
  const float* w_ln2_g  = (const float*)d_in[8];
  const float* w_proj_b = (const float*)d_in[9];
  const float* w_proj_w = (const float*)d_in[10];
  const float* w_qkv_b  = (const float*)d_in[11];
  const float* w_qkv_w  = (const float*)d_in[12];
  const float* w_rpb    = (const float*)d_in[13];

  const float* s_fc1_b  = (const float*)d_in[14];
  const float* s_fc1_w  = (const float*)d_in[15];
  const float* s_fc2_b  = (const float*)d_in[16];
  const float* s_fc2_w  = (const float*)d_in[17];
  const float* s_ln1_b  = (const float*)d_in[18];
  const float* s_ln1_g  = (const float*)d_in[19];
  const float* s_ln2_b  = (const float*)d_in[20];
  const float* s_ln2_g  = (const float*)d_in[21];
  const float* s_proj_b = (const float*)d_in[22];
  const float* s_proj_w = (const float*)d_in[23];
  const float* s_qkv_b  = (const float*)d_in[24];
  const float* s_qkv_w  = (const float*)d_in[25];
  const float* s_rpb    = (const float*)d_in[26];

  const int* rel_idx = (const int*)d_in[27];

  // workspace: transposed f16 weights (both layers), then expanded f16 biases.
  _Float16* ws16 = (_Float16*)d_ws;
  size_t o = 0;
  _Float16* wqkv0  = ws16 + o; o += (size_t)384 * 1152;
  _Float16* wproj0 = ws16 + o; o += (size_t)384 * 384;
  _Float16* w10    = ws16 + o; o += (size_t)384 * 1536;
  _Float16* w20    = ws16 + o; o += (size_t)1536 * 384;
  _Float16* wqkv1  = ws16 + o; o += (size_t)384 * 1152;
  _Float16* wproj1 = ws16 + o; o += (size_t)384 * 384;
  _Float16* w11    = ws16 + o; o += (size_t)384 * 1536;
  _Float16* w21    = ws16 + o; o += (size_t)1536 * 384;
  _Float16* bias0  = ws16 + o; o += (size_t)HH * TT * TT;
  _Float16* bias1  = ws16 + o;

  auto tcvt = [&](const float* s, _Float16* d, int K, int N) {
    int n = K * N;
    transpose_cvt_kernel<<<(n + 255) / 256, 256, 0, stream>>>(s, d, K, N);
  };
  tcvt(w_qkv_w,  wqkv0,  384, 1152);
  tcvt(w_proj_w, wproj0, 384, 384);
  tcvt(w_fc1_w,  w10,    384, 1536);
  tcvt(w_fc2_w,  w20,    1536, 384);
  tcvt(s_qkv_w,  wqkv1,  384, 1152);
  tcvt(s_proj_w, wproj1, 384, 384);
  tcvt(s_fc1_w,  w11,    384, 1536);
  tcvt(s_fc2_w,  w21,    1536, 384);

  const int nb = HH * TT * TT;
  bias_expand_kernel<<<(nb + 255) / 256, 256, 0, stream>>>(w_rpb, rel_idx, bias0);
  bias_expand_kernel<<<(nb + 255) / 256, 256, 0, stream>>>(s_rpb, rel_idx, bias1);

  // d_out = concat(final_x, out1); layer1 writes out1, layer2 reads it.
  float* out_final = (float*)d_out;
  float* out1 = out_final + (size_t)out_size / 2;

  swin3d_block_kernel<<<2048, 256, 0, stream>>>(
      x_in, out1, w_ln1_g, w_ln1_b, w_qkv_b, w_proj_b, w_ln2_g, w_ln2_b,
      w_fc1_b, w_fc2_b, wqkv0, wproj0, w10, w20, bias0, /*shift=*/0);

  swin3d_block_kernel<<<2048, 256, 0, stream>>>(
      out1, out_final, s_ln1_g, s_ln1_b, s_qkv_b, s_proj_b, s_ln2_g, s_ln2_b,
      s_fc1_b, s_fc2_b, wqkv1, wproj1, w11, w21, bias1, /*shift=*/2);
}
